// KGFIT_25357486915919
// MI455X (gfx1250) — compile-verified, matrix operand order checked by reading, not run on hardware
//
#include <hip/hip_runtime.h>
#include <hip/hip_bf16.h>
#include <stdint.h>

// ---- problem constants (from reference) ----
#define K_B     2048
#define K_M     64
#define K_D     512
#define K_NREL  1000
#define K_TWOB  4096
#define K_GAMMA 12.0f
#define K_EPS   1e-12f

typedef __bf16 bf16_t;
typedef __attribute__((ext_vector_type(16))) __bf16        v16bf;
typedef __attribute__((ext_vector_type(8)))  float         v8f;
typedef __attribute__((ext_vector_type(4)))  unsigned int  u32x4;

union FragBF { v16bf v; u32x4 q[2]; };

// Issue one per-lane 16-byte async global->LDS DMA (GVS form: saddr base +
// 32-bit voffset), tracked by ASYNCcnt.  lds_off is a byte offset from the
// wave's LDS base (addrspace(3) pointers are LDS byte offsets).
__device__ inline void async_b128(unsigned lds_off, unsigned gbl_off,
                                  const void* sbase) {
  asm volatile("global_load_async_to_lds_b128 %0, %1, %2"
               :: "v"(lds_off), "v"(gbl_off), "s"(sbase) : "memory");
}
__device__ inline void wait_async0() {
  asm volatile("s_wait_asynccnt 0" ::: "memory");
}

// ---------- block reduction (wave32-aware, blockDim==256) ----------
__device__ inline float block_sum_256(float v, float* s8) {
#pragma unroll
  for (int off = 16; off > 0; off >>= 1) v += __shfl_down(v, off, 32);
  const int lane = threadIdx.x & 31;
  const int w    = threadIdx.x >> 5;
  if (lane == 0) s8[w] = v;
  __syncthreads();
  float r = (threadIdx.x < 8) ? s8[threadIdx.x] : 0.0f;
  if (w == 0) {
#pragma unroll
    for (int off = 4; off > 0; off >>= 1) r += __shfl_down(r, off, 32);
  }
  __syncthreads();
  return r;  // valid on thread 0
}

// ---------- kernel 0: zero accumulators ----------
__global__ void kg_init(float* acc, float* negmean) {
  int t = blockIdx.x * blockDim.x + threadIdx.x;
  if (t < 3)    acc[t] = 0.0f;
  if (t < K_B)  negmean[t] = 0.0f;
}

// ---------- kernel 1: gather + combine + per-row stats ----------
// one block per row i of the 4096-row "clus" matrix (i<2048: head, else tail)
__global__ __launch_bounds__(256) void kg_prep(
    const int* __restrict__ sample, const int* __restrict__ cluster_assign,
    const int* __restrict__ parent_assign,
    const float* __restrict__ rel_emb, const float* __restrict__ e_init,
    const float* __restrict__ e_text, const float* __restrict__ cluster_emb,
    const float* __restrict__ parent_emb,
    bf16_t* __restrict__ clusb, float* __restrict__ sqn,
    float* __restrict__ hr, float* __restrict__ tcomb,
    float* __restrict__ tdist, float* __restrict__ acc) {
  __shared__ float s8[8];
  const int  i    = blockIdx.x;
  const bool head = (i < K_B);
  const int  b    = head ? i : (i - K_B);
  const int  ent  = sample[b * 3 + (head ? 0 : 2)];
  const int  r    = sample[b * 3 + 1] % K_NREL;
  const int  cid  = cluster_assign[ent];
  const int  pid  = parent_assign[cid];
  const float* clu  = cluster_emb + (size_t)cid * K_D;
  const float* par  = parent_emb  + (size_t)pid * K_D;
  const float* ei   = e_init      + (size_t)ent * K_D;
  const float* et   = e_text      + (size_t)ent * K_D;
  const float* rrow = rel_emb     + (size_t)r   * K_D;

  float s_sq = 0.f, s_intra = 0.f, s_par = 0.f, s_text = 0.f;
  for (int d = threadIdx.x; d < K_D; d += 256) {
    const float c  = clu[d];
    const bf16_t cb = (bf16_t)c;           // bf16-rounded value for the WMMA GEMM
    const float cf = (float)cb;
    clusb[(size_t)i * K_D + d] = cb;
    s_sq += cf * cf;                       // sq-norms from the SAME rounded data
    const float comb = 0.5f * ei[d] + 0.5f * et[d];
    const float dc = comb - c;   s_intra += dc * dc;
    const float dp = c - par[d]; s_par   += dp * dp;
    const float dt = comb - et[d]; s_text += dt * dt;
    if (head) hr[(size_t)b * K_D + d]    = comb + rrow[d];
    else      tcomb[(size_t)b * K_D + d] = comb;
  }
  float t_sq    = block_sum_256(s_sq, s8);
  float t_intra = block_sum_256(s_intra, s8);
  float t_par   = block_sum_256(s_par, s8);
  float t_text  = block_sum_256(s_text, s8);
  if (threadIdx.x == 0) {
    sqn[i]   = t_sq;
    tdist[i] = sqrtf(t_text + K_EPS);
    atomicAdd(&acc[0], sqrtf(t_intra + K_EPS));  // intra_cluster sum
    atomicAdd(&acc[2], sqrtf(t_par + K_EPS));    // parent_child sum
  }
}

// ---------- kernel 2: pairwise-L2 row-min via bf16 WMMA ----------
// 64 blocks, 256 threads (8 waves). Block owns 64 rows; sweeps all 4096 cols
// in 32-col tiles. Wave (w&3)=row-subtile, (w>>2)=col-half: a 16x16 C tile.
// Tiles are staged with ASYNCcnt-tracked global->LDS DMA (no VGPR round-trip).
__global__ __launch_bounds__(256) void kg_gemm_min(
    const bf16_t* __restrict__ clusb, const float* __restrict__ sqn,
    float* __restrict__ acc) {
  __shared__ bf16_t lA[64 * K_D];   // 64 KB
  __shared__ bf16_t lB[32 * K_D];   // 32 KB
  __shared__ float  lmin[64 * 32];  // 8 KB
  __shared__ float  s8[8];

  const int i0   = blockIdx.x * 64;
  const int tid  = threadIdx.x;
  const int lane = tid & 31;
  const int wave = tid >> 5;
  const int rowSub = (wave & 3) * 16;   // 0/16/32/48
  const int colOff = (wave >> 2) * 16;  // 0/16 within 32-col tile
  const int laneM  = lane & 15;
  const int kb     = (lane >> 4) * 8;   // ISA 16-bit A/B lane K-base
  const int rowAdd = (lane >> 4) * 8;   // C/D: lanes>=16 hold M+8

  const unsigned ldsA0 = (unsigned)(uintptr_t)&lA[0];
  const unsigned ldsB0 = (unsigned)(uintptr_t)&lB[0];

  // stage A tile (64x512 bf16 = 64 KB) once: 256 lanes x 16 x b128 DMA
  {
    const unsigned baseA = (unsigned)((size_t)i0 * K_D * 2u);
#pragma unroll
    for (int it = 0; it < 16; ++it) {
      const unsigned idx = (unsigned)(tid + it * 256) * 16u;
      async_b128(ldsA0 + idx, baseA + idx, clusb);
    }
  }
  wait_async0();
  __syncthreads();

  float my_sq[8], rmin[8];
#pragma unroll
  for (int e = 0; e < 8; ++e) {
    my_sq[e] = sqn[i0 + rowSub + e + rowAdd];
    rmin[e]  = 3.0e38f;
  }

  for (int j0 = 0; j0 < K_TWOB; j0 += 32) {
    __syncthreads();  // previous tile's fragment reads done
    {
      // stage B tile (32x512 bf16 = 32 KB): 256 lanes x 8 x b128 DMA
      const unsigned baseB = (unsigned)((size_t)j0 * K_D * 2u);
#pragma unroll
      for (int it = 0; it < 8; ++it) {
        const unsigned idx = (unsigned)(tid + it * 256) * 16u;
        async_b128(ldsB0 + idx, baseB + idx, clusb);
      }
    }
    wait_async0();
    __syncthreads();

    v8f c = {0.f, 0.f, 0.f, 0.f, 0.f, 0.f, 0.f, 0.f};
#pragma unroll
    for (int k0 = 0; k0 < K_D; k0 += 32) {
      FragBF a, bm;
      const u32x4* pa = (const u32x4*)(lA + (rowSub + laneM) * K_D + k0 + kb);
      a.q[0] = pa[0];   // K = kb .. kb+7
      a.q[1] = pa[2];   // K = kb+16 .. kb+23
      const u32x4* pb = (const u32x4*)(lB + (colOff + laneM) * K_D + k0 + kb);
      bm.q[0] = pb[0];
      bm.q[1] = pb[2];
      c = __builtin_amdgcn_wmma_f32_16x16x32_bf16(
          false, a.v, false, bm.v, (short)0, c, false, false);
    }

    const int col = j0 + colOff + laneM;
    const float sqj = sqn[col];
#pragma unroll
    for (int e = 0; e < 8; ++e) {
      const int row = i0 + rowSub + e + rowAdd;
      float d2 = my_sq[e] + sqj - 2.0f * c[e];
      d2 = fmaxf(d2, K_EPS);
      float dist = sqrtf(d2);
      if (row == col) dist += 1.0e9f;  // diagonal mask (after sqrt, per ref)
      rmin[e] = fminf(rmin[e], dist);
    }
  }

  __syncthreads();
#pragma unroll
  for (int e = 0; e < 8; ++e) {
    const int rl = rowSub + e + rowAdd;   // 0..63
    const int cs = colOff + laneM;        // 0..31
    lmin[rl * 32 + cs] = rmin[e];
  }
  __syncthreads();

  float part = 0.0f;
  if (tid < 64) {
    float m = lmin[tid * 32];
#pragma unroll
    for (int cc = 1; cc < 32; ++cc) m = fminf(m, lmin[tid * 32 + cc]);
    part = m;
  }
  float tot = block_sum_256(part, s8);
  if (tid == 0) atomicAdd(&acc[1], tot);  // sum of per-row mins
}

// ---------- kernel 3: true_score[b] = GAMMA - ||hr - t_comb||_1 ----------
__global__ __launch_bounds__(256) void kg_true(
    const float* __restrict__ hr, const float* __restrict__ tcomb,
    float* __restrict__ tscore) {
  __shared__ float s8[8];
  const int b = blockIdx.x;
  const float4* a4 = (const float4*)(hr    + (size_t)b * K_D);
  const float4* b4 = (const float4*)(tcomb + (size_t)b * K_D);
  float s = 0.f;
  for (int d = threadIdx.x; d < K_D / 4; d += 256) {
    float4 x = a4[d], y = b4[d];
    s += fabsf(x.x - y.x) + fabsf(x.y - y.y) + fabsf(x.z - y.z) + fabsf(x.w - y.w);
  }
  float t = block_sum_256(s, s8);
  if (threadIdx.x == 0) tscore[b] = K_GAMMA - t;
}

// ---------- kernel 4: negative-sample L1 scores (HBM-gather bound) ----------
// one block per b; hr[b] staged in LDS; 8 waves x 8 negatives each.
// Prefetch next negative's rows (global_prefetch_b8) while reducing current.
__global__ __launch_bounds__(256) void kg_neg(
    const int* __restrict__ neg_tails, const float* __restrict__ e_init,
    const float* __restrict__ e_text, const float* __restrict__ hr,
    float* __restrict__ negmean) {
  __shared__ float sh[K_D];
  const int b = blockIdx.x;
  for (int d = threadIdx.x; d < K_D; d += 256) sh[d] = hr[(size_t)b * K_D + d];
  __syncthreads();

  const int wave = threadIdx.x >> 5;
  const int lane = threadIdx.x & 31;
  const float4* sh4 = (const float4*)sh;
  float accw = 0.f;
#pragma unroll 1
  for (int mi = 0; mi < 8; ++mi) {
    const int m  = wave * 8 + mi;
    const int nt = neg_tails[b * K_M + m];
    if (mi < 7) {  // prefetch next gathered rows into cache
      const int ntn = neg_tails[b * K_M + m + 1];
      __builtin_prefetch(e_init + (size_t)ntn * K_D + lane * 16, 0, 0);
      __builtin_prefetch(e_text + (size_t)ntn * K_D + lane * 16, 0, 0);
    }
    const float4* ei4 = (const float4*)(e_init + (size_t)nt * K_D);
    const float4* et4 = (const float4*)(e_text + (size_t)nt * K_D);
    float s = 0.f;
#pragma unroll
    for (int d4 = lane; d4 < K_D / 4; d4 += 32) {
      float4 a = ei4[d4], t = et4[d4], h = sh4[d4];
      s += fabsf(h.x - (0.5f * a.x + 0.5f * t.x));
      s += fabsf(h.y - (0.5f * a.y + 0.5f * t.y));
      s += fabsf(h.z - (0.5f * a.z + 0.5f * t.z));
      s += fabsf(h.w - (0.5f * a.w + 0.5f * t.w));
    }
#pragma unroll
    for (int off = 16; off > 0; off >>= 1) s += __shfl_down(s, off, 32);
    if (lane == 0) accw += (K_GAMMA - s);
  }
  if (lane == 0) atomicAdd(&negmean[b], accw * (1.0f / (float)K_M));
}

// ---------- kernel 5: final assembly ----------
__global__ void kg_final(const float* __restrict__ acc,
                         const float* __restrict__ tdist,
                         const float* __restrict__ tscore,
                         const float* __restrict__ negmean,
                         float* __restrict__ out) {
  const int b = blockIdx.x * blockDim.x + threadIdx.x;
  if (b >= K_B) return;
  const float inv = 1.0f / (float)K_TWOB;
  const float intra  = acc[0] * inv;
  const float inter  = acc[1] * inv;
  const float parent = acc[2] * inv;
  const float hier = intra - 1.0f * inter + 1.0f * parent;  // LAM1=LAM2=1
  float score = -0.5f * hier - 0.5f * (tdist[b] + tdist[K_B + b]);  // ALPHA=BETA=0.5
  score -= 1.0f * (tscore[b] - negmean[b]);                          // GAMMA_2=1
  out[b] = score;
}

extern "C" void kernel_launch(void* const* d_in, const int* in_sizes, int n_in,
                              void* d_out, int out_size, void* d_ws, size_t ws_size,
                              hipStream_t stream) {
  (void)in_sizes; (void)n_in; (void)out_size; (void)ws_size;
  const int*   sample   = (const int*)d_in[0];
  const int*   neg_tail = (const int*)d_in[1];
  const int*   cl_asg   = (const int*)d_in[2];
  const int*   pa_asg   = (const int*)d_in[3];
  const float* rel      = (const float*)d_in[4];
  const float* einit    = (const float*)d_in[5];
  const float* etext    = (const float*)d_in[6];
  const float* cemb     = (const float*)d_in[7];
  const float* pemb     = (const float*)d_in[8];
  float*       out      = (float*)d_out;

  char* ws = (char*)d_ws;
  bf16_t* clusb   = (bf16_t*)(ws + 0);          // 4096*512*2   = 4,194,304
  float*  sqn     = (float*)(ws + 4194304);     // 4096*4       =    16,384
  float*  hr      = (float*)(ws + 4210688);     // 2048*512*4   = 4,194,304
  float*  tcomb   = (float*)(ws + 8404992);     // 2048*512*4   = 4,194,304
  float*  tdist   = (float*)(ws + 12599296);    // 4096*4
  float*  tscore  = (float*)(ws + 12615680);    // 2048*4
  float*  negmean = (float*)(ws + 12623872);    // 2048*4
  float*  acc     = (float*)(ws + 12632064);    // 3*4

  kg_init<<<9, 256, 0, stream>>>(acc, negmean);
  kg_prep<<<K_TWOB, 256, 0, stream>>>(sample, cl_asg, pa_asg, rel, einit, etext,
                                      cemb, pemb, clusb, sqn, hr, tcomb, tdist, acc);
  kg_gemm_min<<<K_TWOB / 64, 256, 0, stream>>>(clusb, sqn, acc);
  kg_true<<<K_B, 256, 0, stream>>>(hr, tcomb, tscore);
  kg_neg<<<K_B, 256, 0, stream>>>(neg_tail, einit, etext, hr, negmean);
  kg_final<<<(K_B + 255) / 256, 256, 0, stream>>>(acc, tdist, tscore, negmean, out);
}
